// FluxAttnProcessor2_0_85787676770755
// MI455X (gfx1250) — matrix operand order repaired and checked by reading, not compiled, not run on hardware
//
#include <hip/hip_runtime.h>

// ---------------------------------------------------------------------------
// Flux joint attention for MI455X (gfx1250, wave32, WMMA bf16 -> f32).
// B=1, S_IMG=2048, S_TXT=512, D=3072, H=24, HD=128, S = 2560.
// ---------------------------------------------------------------------------

typedef __attribute__((ext_vector_type(16))) __bf16 v16bf;
typedef __attribute__((ext_vector_type(8)))  float  v8f;

#define D3072 3072
#define SEQ   2560
#define STXT  512
#define SIMG  2048
#define NH    24
#define HDIM  128

// ---- WMMA fragment loaders per CDNA5 ISA 7.12.2 (16-bit A 16x32, B 32x16) --
// A: lane holds row m = lane&15; element j (0..15):
//    j<8 : k = 8*(lane>=16) + j          j>=8 : k = 16 + 8*(lane>=16) + (j-8)
__device__ __forceinline__ v16bf frag_a(const __bf16* row, int hi) {
  v16bf f;
  const __bf16* p0 = row + hi * 8;
  const __bf16* p1 = row + 16 + hi * 8;
#pragma unroll
  for (int j = 0; j < 8; ++j) { f[j] = p0[j]; f[j + 8] = p1[j]; }
  return f;
}
// B: lane holds column n = lane&15; element j (same k order as A):
//    j<8 : k = 8*(lane>=16) + j          j>=8 : k = 16 + 8*(lane>=16) + (j-8)
//    `row` must point at B-column n stored contiguously over k (i.e. LDS[n][k]).
__device__ __forceinline__ v16bf frag_b(const __bf16* row, int hi) {
  v16bf f;
  const __bf16* p0 = row + hi * 8;
  const __bf16* p1 = row + 16 + hi * 8;
#pragma unroll
  for (int j = 0; j < 8; ++j) { f[j] = p0[j]; f[j + 8] = p1[j]; }
  return f;
}

// ---------------------------------------------------------------------------
// GEMM: Out[m][n] = sum_k A[m][k] * W[n][k] + bias[n]   (einsum bsd,ed->bse)
// K = ldo = 3072 baked in so all addressing folds to base + immediate offset.
// Region split at row `splitM` (text vs image weights / pointers).
// Block = 256 threads (8 waves), tile 128x128, K-step 32, double-buffered LDS.
// ---------------------------------------------------------------------------
__global__ __launch_bounds__(256)
void gemm_bias_region(const float* __restrict__ A0, const float* __restrict__ A1,
                      const float* __restrict__ W0, const float* __restrict__ W1,
                      const float* __restrict__ B0, const float* __restrict__ B1,
                      float* __restrict__ O0, float* __restrict__ O1,
                      int splitM)
{
  __shared__ __bf16 Al[2][128][32];
  __shared__ __bf16 Bl[2][128][32];

  const int tid  = threadIdx.x;
  const int wv   = tid >> 5;
  const int lane = tid & 31;
  const int hi   = lane >> 4;
  const int nl   = lane & 15;
  const int wm   = wv >> 1;   // 0..3 -> 32-row strip
  const int wn   = wv & 1;    // 0..1 -> 64-col strip

  const int m0 = blockIdx.y * 128;
  const int n0 = blockIdx.x * 128;
  const int region = (m0 >= splitM);
  const float* A  = region ? A1 : A0;
  const float* W  = region ? W1 : W0;
  const float* Bb = region ? B1 : B0;
  float*       O  = region ? O1 : O0;
  const int mb = region ? (m0 - splitM) : m0;

  // per-thread staging coordinates: 1024 float4 per tile / 256 threads = 4 each
  const int sr = tid >> 1;            // used as r = sr + 128*? no: see below
  (void)sr;

  v8f zero = {0.f, 0.f, 0.f, 0.f, 0.f, 0.f, 0.f, 0.f};
  v8f acc[2][4];
#pragma unroll
  for (int i = 0; i < 2; ++i)
#pragma unroll
    for (int j = 0; j < 4; ++j) acc[i][j] = zero;

  // ---- prologue: stage k0 = 0 into buffer 0 ----
#pragma unroll
  for (int l = 0; l < 4; ++l) {
    int u = tid + 256 * l;            // 0..1023
    int r  = u >> 3;                  // 0..127
    int kq = (u & 7) * 4;             // 0..28
    float4 av = *(const float4*)(A + (size_t)(mb + r) * D3072 + kq);
    Al[0][r][kq + 0] = (__bf16)av.x; Al[0][r][kq + 1] = (__bf16)av.y;
    Al[0][r][kq + 2] = (__bf16)av.z; Al[0][r][kq + 3] = (__bf16)av.w;
    float4 wv4 = *(const float4*)(W + (size_t)(n0 + r) * D3072 + kq);
    Bl[0][r][kq + 0] = (__bf16)wv4.x; Bl[0][r][kq + 1] = (__bf16)wv4.y;
    Bl[0][r][kq + 2] = (__bf16)wv4.z; Bl[0][r][kq + 3] = (__bf16)wv4.w;
  }
  __syncthreads();

  const int NK = D3072 / 32;          // 96 k-steps
  for (int it = 0; it < NK; ++it) {
    const int cur = it & 1;
    const bool more = (it + 1 < NK);

    // issue next tile's global loads into registers (overlaps with WMMA below)
    float4 ar[4], wr[4];
    if (more) {
      const int k0 = (it + 1) * 32;
#pragma unroll
      for (int l = 0; l < 4; ++l) {
        int u = tid + 256 * l;
        int r  = u >> 3;
        int kq = (u & 7) * 4;
        ar[l] = *(const float4*)(A + (size_t)(mb + r) * D3072 + k0 + kq);
        wr[l] = *(const float4*)(W + (size_t)(n0 + r) * D3072 + k0 + kq);
      }
      if (it + 2 < NK) {  // prefetch the tile after that into L2/L0
        int u = tid;      // one line per thread is plenty
        int r = u >> 3, kq = (u & 7) * 4;
        __builtin_prefetch(A + (size_t)(mb + r) * D3072 + (it + 2) * 32 + kq, 0, 3);
        __builtin_prefetch(W + (size_t)(n0 + r) * D3072 + (it + 2) * 32 + kq, 0, 3);
      }
    }

    // compute on current buffer
    v16bf a[2], b[4];
#pragma unroll
    for (int i = 0; i < 2; ++i) a[i] = frag_a(&Al[cur][wm * 32 + i * 16 + nl][0], hi);
#pragma unroll
    for (int j = 0; j < 4; ++j) b[j] = frag_b(&Bl[cur][wn * 64 + j * 16 + nl][0], hi);
#pragma unroll
    for (int i = 0; i < 2; ++i)
#pragma unroll
      for (int j = 0; j < 4; ++j)
        acc[i][j] = __builtin_amdgcn_wmma_f32_16x16x32_bf16(
            false, a[i], false, b[j], (short)0, acc[i][j], false, false);

    // convert + store next tile into the other buffer
    if (more) {
      const int nxt = cur ^ 1;
#pragma unroll
      for (int l = 0; l < 4; ++l) {
        int u = tid + 256 * l;
        int r  = u >> 3;
        int kq = (u & 7) * 4;
        Al[nxt][r][kq + 0] = (__bf16)ar[l].x; Al[nxt][r][kq + 1] = (__bf16)ar[l].y;
        Al[nxt][r][kq + 2] = (__bf16)ar[l].z; Al[nxt][r][kq + 3] = (__bf16)ar[l].w;
        Bl[nxt][r][kq + 0] = (__bf16)wr[l].x; Bl[nxt][r][kq + 1] = (__bf16)wr[l].y;
        Bl[nxt][r][kq + 2] = (__bf16)wr[l].z; Bl[nxt][r][kq + 3] = (__bf16)wr[l].w;
      }
    }
    __syncthreads();
  }

  // epilogue: bias + store (C layout: m = v + 8*hi, n = lane&15)
#pragma unroll
  for (int i = 0; i < 2; ++i) {
#pragma unroll
    for (int j = 0; j < 4; ++j) {
      int n = n0 + wn * 64 + j * 16 + nl;
      float bias = Bb[n];
      float* p = O + (size_t)(mb + wm * 32 + i * 16 + 8 * hi) * D3072 + n;
#pragma unroll
      for (int v = 0; v < 8; ++v)
        p[v * D3072] = acc[i][j][v] + bias;   // immediate-offset stores
    }
  }
}

// ---------------------------------------------------------------------------
// Per-(row, head) RMSNorm + RoPE, fused in place on Q and K.
// One wave handles one 128-wide head row (4 floats / lane).
// ---------------------------------------------------------------------------
__global__ __launch_bounds__(256)
void rmsrope_kernel(float* __restrict__ Q, float* __restrict__ Kb,
                    const float* __restrict__ rope_cos, const float* __restrict__ rope_sin,
                    const float* __restrict__ nqw, const float* __restrict__ nkw,
                    const float* __restrict__ naqw, const float* __restrict__ nakw)
{
  const int idx  = blockIdx.x * 8 + (threadIdx.x >> 5);
  const int lane = threadIdx.x & 31;
  const int ROWS = SEQ * NH;
  const int which = idx / ROWS;            // 0 = Q, 1 = K
  const int rem   = idx - which * ROWS;
  const int s = rem / NH;
  const int h = rem - (rem / NH) * NH;

  float* buf = which ? Kb : Q;
  const float* nw = which ? (s < STXT ? nakw : nkw)
                          : (s < STXT ? naqw : nqw);

  float* p = buf + (size_t)s * D3072 + h * HDIM + lane * 4;
  float4 x = *(float4*)p;
  float ss = x.x * x.x + x.y * x.y + x.z * x.z + x.w * x.w;
#pragma unroll
  for (int m = 16; m >= 1; m >>= 1) ss += __shfl_xor(ss, m, 32);
  float r = rsqrtf(ss * (1.0f / HDIM) + 1e-6f);

  float4 w4 = *(const float4*)(nw + lane * 4);
  float4 y;
  y.x = x.x * r * w4.x; y.y = x.y * r * w4.y;
  y.z = x.z * r * w4.z; y.w = x.w * r * w4.w;

  float4 c  = *(const float4*)(rope_cos + (size_t)s * HDIM + lane * 4);
  float4 sn = *(const float4*)(rope_sin + (size_t)s * HDIM + lane * 4);
  float4 o;
  o.x = y.x * c.x - y.y * sn.x;   // even:  x*cos - x_odd*sin
  o.y = y.y * c.y + y.x * sn.y;   // odd :  x*cos + x_even*sin
  o.z = y.z * c.z - y.w * sn.z;
  o.w = y.w * c.w + y.z * sn.w;
  *(float4*)p = o;
}

// ---------------------------------------------------------------------------
// Flash attention: grid (S/64, H), block 128 (4 waves x 16 Q rows).
// K/V tiles of 32 keys staged to LDS (V transposed), streaming softmax.
// ---------------------------------------------------------------------------
__global__ __launch_bounds__(128)
void flash_attn_kernel(const float* __restrict__ Q, const float* __restrict__ Kb,
                       const float* __restrict__ V, float* __restrict__ O)
{
  __shared__ __bf16 Ql[64][HDIM];     // [q row][hd]
  __shared__ __bf16 Kl[32][HDIM];     // [key][hd]   -> B frags for Q*K^T
  __shared__ __bf16 Vt[HDIM][32];     // [hd][key]   -> B frags for P*V
  __shared__ __bf16 Pl[4][16][32];    // per-wave P tile (C -> A relayout)

  const int tid = threadIdx.x, wv = tid >> 5, lane = tid & 31;
  const int hi = lane >> 4, nl = lane & 15;
  const int q0 = blockIdx.x * 64;
  const int h  = blockIdx.y;
  const size_t cbase = (size_t)h * HDIM;

  // stage this block's 64 Q rows (fp32 -> bf16)
#pragma unroll
  for (int l = 0; l < 16; ++l) {
    int u = tid + 128 * l;
    int m = u >> 5, q4 = (u & 31) * 4;
    float4 v4 = *(const float4*)(Q + (size_t)(q0 + m) * D3072 + cbase + q4);
    Ql[m][q4 + 0] = (__bf16)v4.x; Ql[m][q4 + 1] = (__bf16)v4.y;
    Ql[m][q4 + 2] = (__bf16)v4.z; Ql[m][q4 + 3] = (__bf16)v4.w;
  }
  __syncthreads();

  v16bf aq[4];
#pragma unroll
  for (int c = 0; c < 4; ++c) aq[c] = frag_a(&Ql[wv * 16 + nl][c * 32], hi);

  v8f zero = {0.f, 0.f, 0.f, 0.f, 0.f, 0.f, 0.f, 0.f};
  v8f o[8];
#pragma unroll
  for (int c = 0; c < 8; ++c) o[c] = zero;
  float rm[8], rl[8];
#pragma unroll
  for (int v = 0; v < 8; ++v) { rm[v] = -3.0e38f; rl[v] = 0.f; }
  const float scale = 0.08838834764831845f;   // 1/sqrt(128)

  // per-thread staging coords for K/V tiles
  const int key0 = tid >> 5;          // l-step adds 4
  const int qk4  = (tid & 31) * 4;

  for (int kb = 0; kb < SEQ; kb += 32) {
    __syncthreads();   // previous iteration's LDS consumers done
    if (kb + 32 < SEQ) {  // prefetch next K/V tiles (global_prefetch_b8)
      __builtin_prefetch(Kb + (size_t)(kb + 32 + key0) * D3072 + cbase + qk4, 0, 3);
      __builtin_prefetch(V  + (size_t)(kb + 32 + key0) * D3072 + cbase + qk4, 0, 3);
    }
#pragma unroll
    for (int l = 0; l < 8; ++l) {
      int key = key0 + 4 * l;
      float4 kv = *(const float4*)(Kb + (size_t)(kb + key) * D3072 + cbase + qk4);
      Kl[key][qk4 + 0] = (__bf16)kv.x; Kl[key][qk4 + 1] = (__bf16)kv.y;
      Kl[key][qk4 + 2] = (__bf16)kv.z; Kl[key][qk4 + 3] = (__bf16)kv.w;
      float4 vv = *(const float4*)(V + (size_t)(kb + key) * D3072 + cbase + qk4);
      Vt[qk4 + 0][key] = (__bf16)vv.x; Vt[qk4 + 1][key] = (__bf16)vv.y;
      Vt[qk4 + 2][key] = (__bf16)vv.z; Vt[qk4 + 3][key] = (__bf16)vv.w;
    }
    __syncthreads();

    // scores: 16 q-rows x 32 keys, two 16x16 C tiles, K-dim 128 = 4 steps
    v8f sc[2]; sc[0] = zero; sc[1] = zero;
#pragma unroll
    for (int nc = 0; nc < 2; ++nc)
#pragma unroll
      for (int kc = 0; kc < 4; ++kc) {
        v16bf bk = frag_b(&Kl[nc * 16 + nl][kc * 32], hi);
        sc[nc] = __builtin_amdgcn_wmma_f32_16x16x32_bf16(
            false, aq[kc], false, bk, (short)0, sc[nc], false, false);
      }

    // streaming softmax: lane reduces its 8 rows across its 16-lane half
#pragma unroll
    for (int v = 0; v < 8; ++v) {
      float s0 = sc[0][v] * scale, s1 = sc[1][v] * scale;
      float tm = fmaxf(s0, s1);
#pragma unroll
      for (int m = 8; m >= 1; m >>= 1) tm = fmaxf(tm, __shfl_xor(tm, m, 32));
      float nm = fmaxf(rm[v], tm);
      float alpha = __expf(rm[v] - nm);
      float p0 = __expf(s0 - nm), p1 = __expf(s1 - nm);
      float ps = p0 + p1;
#pragma unroll
      for (int m = 8; m >= 1; m >>= 1) ps += __shfl_xor(ps, m, 32);
      rl[v] = rl[v] * alpha + ps;
      rm[v] = nm;
#pragma unroll
      for (int c = 0; c < 8; ++c) o[c][v] *= alpha;
      int mrow = v + 8 * hi;
      Pl[wv][mrow][nl]      = (__bf16)p0;
      Pl[wv][mrow][16 + nl] = (__bf16)p1;
    }
    __syncthreads();   // P visible (C layout -> A layout relayout via LDS)

    // P (16x32) x V (32x128): 8 WMMAs over hd chunks
    v16bf ap = frag_a(&Pl[wv][nl][0], hi);
#pragma unroll
    for (int c = 0; c < 8; ++c) {
      v16bf bv = frag_b(&Vt[c * 16 + nl][0], hi);
      o[c] = __builtin_amdgcn_wmma_f32_16x16x32_bf16(
          false, ap, false, bv, (short)0, o[c], false, false);
    }
  }

  // normalize and store attn output in (s, h*HD + hd) layout
#pragma unroll
  for (int v = 0; v < 8; ++v) {
    float inv = 1.0f / rl[v];
    float* orow = O + (size_t)(q0 + wv * 16 + v + 8 * hi) * D3072 + cbase;
#pragma unroll
    for (int c = 0; c < 8; ++c)
      orow[c * 16 + nl] = o[c][v] * inv;
  }
}

// ---------------------------------------------------------------------------
extern "C" void kernel_launch(void* const* d_in, const int* in_sizes, int n_in,
                              void* d_out, int out_size, void* d_ws, size_t ws_size,
                              hipStream_t stream)
{
  (void)in_sizes; (void)n_in; (void)out_size; (void)ws_size;
  const float* hid  = (const float*)d_in[0];
  const float* enc  = (const float*)d_in[1];
  const float* rcos = (const float*)d_in[2];
  const float* rsin = (const float*)d_in[3];
  const float* wq   = (const float*)d_in[4];  const float* bq  = (const float*)d_in[5];
  const float* wk   = (const float*)d_in[6];  const float* bk  = (const float*)d_in[7];
  const float* wvm  = (const float*)d_in[8];  const float* bv  = (const float*)d_in[9];
  const float* waq  = (const float*)d_in[10]; const float* baq = (const float*)d_in[11];
  const float* wak  = (const float*)d_in[12]; const float* bak = (const float*)d_in[13];
  const float* wav  = (const float*)d_in[14]; const float* bav = (const float*)d_in[15];
  const float* nqw  = (const float*)d_in[16]; const float* nkw = (const float*)d_in[17];
  const float* naqw = (const float*)d_in[18]; const float* nakw= (const float*)d_in[19];
  const float* wo   = (const float*)d_in[20]; const float* bo  = (const float*)d_in[21];
  const float* wao  = (const float*)d_in[22]; const float* bao = (const float*)d_in[23];
  float* out = (float*)d_out;

  const size_t SD = (size_t)SEQ * D3072;
  float* Qb = (float*)d_ws;
  float* Kb = Qb + SD;
  float* Vb = Kb + SD;
  float* AO = Vb + SD;

  dim3 ggrid(D3072 / 128, SEQ / 128), gblk(256);

  // fused enc+img QKV projections (rows [0,512) = text, [512,2560) = image)
  gemm_bias_region<<<ggrid, gblk, 0, stream>>>(
      enc, hid, waq, wq, baq, bq, Qb, Qb + (size_t)STXT * D3072, STXT);
  gemm_bias_region<<<ggrid, gblk, 0, stream>>>(
      enc, hid, wak, wk, bak, bk, Kb, Kb + (size_t)STXT * D3072, STXT);
  gemm_bias_region<<<ggrid, gblk, 0, stream>>>(
      enc, hid, wav, wvm, bav, bv, Vb, Vb + (size_t)STXT * D3072, STXT);

  // RMSNorm (per head, per region) + RoPE on Q and K in place
  rmsrope_kernel<<<dim3(2 * SEQ * NH / 8), dim3(256), 0, stream>>>(
      Qb, Kb, rcos, rsin, nqw, nkw, naqw, nakw);

  // flash attention -> AO in (s, h*HD) layout
  flash_attn_kernel<<<dim3(SEQ / 64, NH), dim3(128), 0, stream>>>(Qb, Kb, Vb, AO);

  // dual output projection: img rows -> img_out (d_out[0..]), enc rows -> enc_out
  gemm_bias_region<<<ggrid, gblk, 0, stream>>>(
      AO, AO + (size_t)STXT * D3072, wao, wo, bao, bo,
      out + (size_t)SIMG * D3072, out, STXT);
}
